// Topo_Attention_89945205113504
// MI455X (gfx1250) — matrix operand order repaired and checked by reading, compiled, not run on hardware
//
#include <hip/hip_runtime.h>

typedef __bf16 bf16_t;
typedef __attribute__((ext_vector_type(16))) __bf16 v16bf;
typedef __attribute__((ext_vector_type(8)))  __bf16 v8bf;
typedef __attribute__((ext_vector_type(4)))  __bf16 v4bf;
typedef __attribute__((ext_vector_type(8)))  float  v8f;
typedef __attribute__((ext_vector_type(4)))  float  v4f;

// Problem constants (from reference): B=32, T=4096, M=256, L_TP=128
#define TT 4096
#define MM 256
#define LL 128

__device__ __forceinline__ v16bf cvt_frag(v4f f0, v4f f1, v4f f2, v4f f3) {
  v4bf c0 = __builtin_convertvector(f0, v4bf);
  v4bf c1 = __builtin_convertvector(f1, v4bf);
  v4bf c2 = __builtin_convertvector(f2, v4bf);
  v4bf c3 = __builtin_convertvector(f3, v4bf);
  v8bf lo = __builtin_shufflevector(c0, c1, 0, 1, 2, 3, 4, 5, 6, 7);
  v8bf hi = __builtin_shufflevector(c2, c3, 0, 1, 2, 3, 4, 5, 6, 7);
  return __builtin_shufflevector(lo, hi, 0, 1, 2, 3, 4, 5, 6, 7,
                                 8, 9, 10, 11, 12, 13, 14, 15);
}

// Hardware tanh when the gfx1250 builtin exists; exp-based fallback otherwise.
__device__ __forceinline__ float fast_tanh(float x) {
#if __has_builtin(__builtin_amdgcn_tanhf)
  return __builtin_amdgcn_tanhf(x);
#elif __has_builtin(__builtin_amdgcn_tanh_f32)
  return __builtin_amdgcn_tanh_f32(x);
#else
  return 1.f - 2.f / (__expf(2.f * x) + 1.f);
#endif
}

// min 4 waves/EU -> VGPR cap 256 (no VGPR-MSB churn, decent occupancy)
__global__ __launch_bounds__(256, 4) void topo_attn_wmma(
    const float* __restrict__ h1,   // (B, T, M)
    const float* __restrict__ h2,   // (B, T, M)
    const float* __restrict__ w_tp, // (1, L)
    const float* __restrict__ v_tp, // (L, M)
    float* __restrict__ out)        // (B, 2, T)
{
  // v_tp staged as bf16, row-major 128x256, 16B chunks XOR-swizzled by (row&15)
  __shared__ bf16_t vlds[LL * MM]; // 64 KB

  const int tid  = threadIdx.x;
  const int lane = tid & 31;
  const int wave = tid >> 5;
  const int half = lane >> 4;   // which 16-lane half
  const int lrow = lane & 15;   // M-row for A frag / N-col for B,D frags

  // ---- stage v_tp (f32 global) -> bf16 LDS, swizzled ----
  {
    const int row  = tid >> 1;          // 0..127
    const int col0 = (tid & 1) * 128;   // 0 or 128
    const int sw   = row & 15;
    const float* src = v_tp + row * MM + col0;
#pragma unroll
    for (int j = 0; j < 128; j += 8) {
      v4f a = *(const v4f*)(src + j);
      v4f b = *(const v4f*)(src + j + 4);
      v4bf ca = __builtin_convertvector(a, v4bf);
      v4bf cb = __builtin_convertvector(b, v4bf);
      v8bf pk = __builtin_shufflevector(ca, cb, 0, 1, 2, 3, 4, 5, 6, 7);
      const int chunk = (col0 + j) >> 3;               // 16B chunk index 0..31
      *(v8bf*)&vlds[row * MM + ((chunk ^ sw) << 3)] = pk;
    }
  }
  __syncthreads();

  // per-lane slice of w_tp: wv[nt] = w[nt*16 + lrow]
  float wv[8];
#pragma unroll
  for (int nt = 0; nt < 8; ++nt) wv[nt] = w_tp[nt * 16 + lrow];

  const int  tile = blockIdx.x * 8 + wave;   // one 16-row (b,t) tile per wave
  const int  bt0  = tile * 16;
  const long arow = (long)(bt0 + lrow) * MM; // this lane's A-matrix row

  // Fused dual-stream GEMM: share every B-fragment between h1 and h2
  v8f acc1[8] = {};
  v8f acc2[8] = {};

#pragma unroll
  for (int ks = 0; ks < 8; ++ks) {
    const int kb = ks * 32;
    // ---- A fragments: 16-bit A 16x32 layout (interleaved K halves) ----
    // lane(half,lrow): e0..7 -> K = kb+8*half+(0..7); e8..15 -> K = kb+16+8*half+(0..7)
    const float* pa = h1 + arow + kb + 8 * half;
    const float* pb = h2 + arow + kb + 8 * half;
    v16bf a1 = cvt_frag(*(const v4f*)pa,        *(const v4f*)(pa + 4),
                        *(const v4f*)(pa + 16), *(const v4f*)(pa + 20));
    v16bf a2 = cvt_frag(*(const v4f*)pb,        *(const v4f*)(pb + 4),
                        *(const v4f*)(pb + 16), *(const v4f*)(pb + 20));

    // ---- batch ALL B-fragment LDS loads for this K-step (distinct regs ->
    //      one ds clause + single wait, then a 16-WMMA burst) ----
    v8bf bx0[8], bx1[8];
#pragma unroll
    for (int nt = 0; nt < 8; ++nt) {
      // column = nt*16+lrow, K = kb+8*half+(0..7) and kb+16+8*half+(0..7); swizzle key = lrow
      const int brow = nt * 16 + lrow;
      const int cb0  = (kb + 8 * half) >> 3;
      bx0[nt] = *(const v8bf*)&vlds[brow * MM + (((cb0    ) ^ lrow) << 3)];
      bx1[nt] = *(const v8bf*)&vlds[brow * MM + (((cb0 + 2) ^ lrow) << 3)];
    }

#pragma unroll
    for (int nt = 0; nt < 8; ++nt) {
      v16bf bfm = __builtin_shufflevector(bx0[nt], bx1[nt],
                                          0, 1, 2, 3, 4, 5, 6, 7,
                                          8, 9, 10, 11, 12, 13, 14, 15);
      acc1[nt] = __builtin_amdgcn_wmma_f32_16x16x32_bf16(
          false, a1, false, bfm, (short)0, acc1[nt], false, false);
      acc2[nt] = __builtin_amdgcn_wmma_f32_16x16x32_bf16(
          false, a2, false, bfm, (short)0, acc2[nt], false, false);
    }
  }

  // epilogue: p[r] = sum_nt wv[nt] * tanh(acc[nt][r])
  // D layout: lane(half,lrow), vgpr r -> D[m=r+8*half][n=nt*16+lrow]
  float p1[8], p2[8];
#pragma unroll
  for (int r = 0; r < 8; ++r) { p1[r] = 0.f; p2[r] = 0.f; }
#pragma unroll
  for (int nt = 0; nt < 8; ++nt) {
#pragma unroll
    for (int r = 0; r < 8; ++r) {
      p1[r] += wv[nt] * fast_tanh(acc1[nt][r]);
      p2[r] += wv[nt] * fast_tanh(acc2[nt][r]);
    }
  }

  // reduce over the 16 lanes of each half (the N=L dimension)
#pragma unroll
  for (int msk = 1; msk < 16; msk <<= 1) {
#pragma unroll
    for (int r = 0; r < 8; ++r) {
      p1[r] += __shfl_xor(p1[r], msk, 32);
      p2[r] += __shfl_xor(p2[r], msk, 32);
    }
  }

  // lanes 0..7 write m=0..7 (half 0); lanes 16..23 write m=8..15 (half 1)
  if (lrow < 8) {
    const int r = lrow;
    auto pick = [&](const float* p) -> float {
      float u0 = (r & 1) ? p[1] : p[0];
      float u1 = (r & 1) ? p[3] : p[2];
      float u2 = (r & 1) ? p[5] : p[4];
      float u3 = (r & 1) ? p[7] : p[6];
      float q0 = (r & 2) ? u1 : u0;
      float q1 = (r & 2) ? u3 : u2;
      return (r & 4) ? q1 : q0;
    };
    const float s1v = pick(p1);
    const float s2v = pick(p2);
    const int   m   = r + 8 * half;
    const int   bt  = bt0 + m;
    const int   b   = bt >> 12;          // T = 4096
    const int   t   = bt & (TT - 1);
    // exact 2-way softmax
    const float a1 = 1.f / (1.f + __expf(s2v - s1v));
    const float a2 = 1.f / (1.f + __expf(s1v - s2v));
    out[(long)(b * 2 + 0) * TT + t] = a1;
    out[(long)(b * 2 + 1) * TT + t] = a2;
  }
}

extern "C" void kernel_launch(void* const* d_in, const int* in_sizes, int n_in,
                              void* d_out, int out_size, void* d_ws, size_t ws_size,
                              hipStream_t stream) {
  (void)in_sizes; (void)n_in; (void)out_size; (void)d_ws; (void)ws_size;
  const float* h1 = (const float*)d_in[0];
  const float* h2 = (const float*)d_in[1];
  const float* w  = (const float*)d_in[2];
  const float* v  = (const float*)d_in[3];
  float* out = (float*)d_out;

  // B*T/16 row-tiles = 8192 waves; 8 waves (256 threads) per block -> 1024 blocks
  topo_attn_wmma<<<dim3(1024), dim3(256), 0, stream>>>(h1, h2, w, v, out);
}
